// BiLSTMTranslator_36627481100506
// MI455X (gfx1250) — matrix operand order repaired and checked by reading, not compiled, not run on hardware
//
#include <hip/hip_runtime.h>
#include <hip/hip_bf16.h>
#include <math.h>

// ---------------------------------------------------------------------------
// BiLSTM translator forward for MI455X (gfx1250, wave32, WMMA + TDM).
// Matrix math: v_wmma_f32_16x16x32_f16 (f16 in, f32 acc), 4 WMMAs per wave
// per K-step (32x32 register block). Tile movement: Tensor Data Mover
// (tensor_load_to_lds) into double-buffered pad-striped LDS, paced with
// s_wait_tensorcnt; fragments read from LDS (ds_load_b64).
// ---------------------------------------------------------------------------

typedef __attribute__((ext_vector_type(16))) _Float16 v16h;
typedef __attribute__((ext_vector_type(8)))  float    v8f;
typedef unsigned int v4u __attribute__((ext_vector_type(4)));
typedef int          v4i __attribute__((ext_vector_type(4)));
typedef int          v8i __attribute__((ext_vector_type(8)));

// Problem constants (match reference).
#define Bb   32
#define Ss   64
#define Tt   48
#define VT_  32000
#define Ee   256
#define Hh   512
#define H2_  1024
#define Gg   (4*Hh)    // 2048
#define G2_  (4*H2_)   // 4096

// LDS tile pitch: 32 K-dwords + 2 pad dwords (TDM pad_interval=32dw, amount=2dw)
#define LDSPITCH 34

// ---------------------------------------------------------------------------
// Issue one TDM 2D tile load: `rows` rows of 32 f32 each, row stride strideK
// (in f32 units), from gptr into LDS at lds_byte_off, padded to LDSPITCH.
// Executed by a single wave; tracked with TENSORcnt.
// ---------------------------------------------------------------------------
__device__ __forceinline__ void tdm_load_tile_f32(const float* gptr,
                                                  unsigned lds_byte_off,
                                                  int rows, int strideK) {
  unsigned long long ga = (unsigned long long)(uintptr_t)gptr;
  v4u g0;
  g0[0] = 1u;                                              // count=1, user mode
  g0[1] = lds_byte_off;                                    // lds_addr (bytes)
  g0[2] = (unsigned)ga;                                    // global_addr[31:0]
  g0[3] = (unsigned)((ga >> 32) & 0x01FFFFFFull) | (2u << 30); // addr[56:32] | type=2
  v8i g1;
  // data_size=4B (2<<16), pad_enable (1<<20), pad_interval=32dw (4<<22),
  // pad_amount=2dw (1<<25); workgroup_mask=0 (not in cluster)
  g1[0] = (2 << 16) | (1 << 20) | (4 << 22) | (1 << 25);
  g1[1] = (int)(32u << 16);                                // tensor_dim0[15:0]=32
  g1[2] = (int)((unsigned)(rows & 0xFFFF) << 16);          // tensor_dim1[15:0]=rows
  g1[3] = (int)(32u << 16);                                // tile_dim0=32
  g1[4] = rows & 0xFFFF;                                   // tile_dim1=rows, tile_dim2=0
  g1[5] = strideK;                                         // tensor_dim0_stride[31:0]
  g1[6] = 0;
  g1[7] = 0;
  v4i z4 = {};
#if defined(__clang_major__) && __clang_major__ >= 23
  v8i z8 = {};
  __builtin_amdgcn_tensor_load_to_lds(g0, g1, z4, z4, z8, 0);
#else
  __builtin_amdgcn_tensor_load_to_lds(g0, g1, z4, z4, 0);
#endif
}

// ---------------------------------------------------------------------------
// Generic WMMA GEMM:  Cout[row][n] = sum_k A[row][k]*W[n][k] + bias[n] + Cin[row][n]
//   A : M x K f32 row-major (M%32==0), W : N x K f32 row-major (N%128==0), K%32==0.
// permMod > 0 remaps output rows: orow = (row%permMod)*(M/permMod) + row/permMod
// (writes (t,b) rows directly into (b,t,VT) output order).
// Block = 128 threads = 4 waves; each wave computes a 32x32 C block
// (2 M-tiles x 2 N-tiles = 4 WMMA accumulators, A/B fragments reused 2x).
// Block tile = 32 (M) x 128 (N); grid = (N/128, M/32).
// K-loop: TDM streams A(32x32) and W(128x32) f32 tiles into double-buffered
// LDS; waves build f16 fragments from LDS and issue 4 WMMAs per K-step.
// ---------------------------------------------------------------------------
__global__ __launch_bounds__(128)
void gemm_wmma_f16(const float* __restrict__ A, const float* __restrict__ W,
                   const float* __restrict__ bias, const float* __restrict__ Cin,
                   float* __restrict__ Cout, int M, int N, int K, int permMod) {
  __shared__ float As[2][32  * LDSPITCH];
  __shared__ float Bs[2][128 * LDSPITCH];

  const int lane = threadIdx.x & 31;
  const int w    = threadIdx.x >> 5;          // 0..3 (wave's N strip)
  const int hi   = lane >> 4;                 // 0 or 1
  const int lo   = lane & 15;

  const int m0b = blockIdx.y * 32;
  const int n0b = blockIdx.x * 128;
  const bool issuer = (threadIdx.x < 32);     // wave 0 drives the TDM

  // Prologue: fetch first K-slice.
  if (issuer) {
    tdm_load_tile_f32(A + (size_t)m0b * K, (unsigned)(uintptr_t)&As[0][0], 32, K);
    tdm_load_tile_f32(W + (size_t)n0b * K, (unsigned)(uintptr_t)&Bs[0][0], 128, K);
    __builtin_amdgcn_s_wait_tensorcnt(0);
  }
  __syncthreads();

  v8f acc00 = {}, acc01 = {}, acc10 = {}, acc11 = {};
  int cur = 0;
  for (int kb = 0; kb < K; kb += 32, cur ^= 1) {
    const bool has_next = (kb + 32) < K;
    // Kick off next slice into the other buffer (overlaps with compute below).
    if (issuer && has_next) {
      tdm_load_tile_f32(A + (size_t)m0b * K + kb + 32,
                        (unsigned)(uintptr_t)&As[cur ^ 1][0], 32, K);
      tdm_load_tile_f32(W + (size_t)n0b * K + kb + 32,
                        (unsigned)(uintptr_t)&Bs[cur ^ 1][0], 128, K);
    }

    const float* __restrict__ Ab = &As[cur][0];
    const float* __restrict__ Wb = &Bs[cur][0];
    v16h af0, af1, bf0, bf1;
    // A fragment: 16x32 f16 layout (ISA 7.12.2):
    //   VGPR v, half p -> K = (v<4?0:16) + 2*(v&3) + 8*hi + p
    // B fragment: 32x16 f16 layout: VGPR v, half p -> K = (v<4?0:16) + 2*(v&3) + 8*hi + p
#pragma unroll
    for (int v = 0; v < 8; ++v) {
      const int ka = ((v & 4) ? 16 : 0) + 2 * (v & 3) + 8 * hi;
      const float2 a0 = *(const float2*)(Ab + (lo)      * LDSPITCH + ka);
      const float2 a1 = *(const float2*)(Ab + (16 + lo) * LDSPITCH + ka);
      af0[2 * v] = (_Float16)a0.x; af0[2 * v + 1] = (_Float16)a0.y;
      af1[2 * v] = (_Float16)a1.x; af1[2 * v + 1] = (_Float16)a1.y;
      const int kv = ((v & 4) ? 16 : 0) + 2 * (v & 3) + 8 * hi;
      const float2 b0 = *(const float2*)(Wb + (w * 32 + lo)      * LDSPITCH + kv);
      const float2 b1 = *(const float2*)(Wb + (w * 32 + 16 + lo) * LDSPITCH + kv);
      bf0[2 * v] = (_Float16)b0.x; bf0[2 * v + 1] = (_Float16)b0.y;
      bf1[2 * v] = (_Float16)b1.x; bf1[2 * v + 1] = (_Float16)b1.y;
    }
    acc00 = __builtin_amdgcn_wmma_f32_16x16x32_f16(false, af0, false, bf0, (short)0, acc00, false, false);
    acc01 = __builtin_amdgcn_wmma_f32_16x16x32_f16(false, af0, false, bf1, (short)0, acc01, false, false);
    acc10 = __builtin_amdgcn_wmma_f32_16x16x32_f16(false, af1, false, bf0, (short)0, acc10, false, false);
    acc11 = __builtin_amdgcn_wmma_f32_16x16x32_f16(false, af1, false, bf1, (short)0, acc11, false, false);

    if (issuer && has_next) __builtin_amdgcn_s_wait_tensorcnt(0);
    __syncthreads();
  }

  // Epilogue: bias + optional addend + (optional) row permutation, 4 tiles.
  const int Tq = (permMod > 0) ? (M / permMod) : 0;
  const v8f* accs[4] = { &acc00, &acc01, &acc10, &acc11 };
#pragma unroll
  for (int mi = 0; mi < 2; ++mi) {
#pragma unroll
    for (int ni = 0; ni < 2; ++ni) {
      const v8f& acc = *accs[mi * 2 + ni];
      const int m0 = m0b + mi * 16;
      const int bn = n0b + w * 32 + ni * 16 + lo;
      const float bv = bias ? bias[bn] : 0.0f;
#pragma unroll
      for (int r = 0; r < 8; ++r) {
        const int row = m0 + r + 8 * hi;        // C layout: VGPR r -> M = r + 8*hi
        float v = acc[r] + bv;
        if (Cin) v += Cin[(size_t)row * N + bn];
        size_t orow = (size_t)row;
        if (permMod > 0) orow = (size_t)(row % permMod) * Tq + (row / permMod);
        Cout[orow * (size_t)N + bn] = v;
      }
    }
  }
}

// ---------------------------------------------------------------------------
// Embedding gather into time-major layout: out[(l*B + b)*E + e] = emb[tok[b*L+l]*E + e]
// ---------------------------------------------------------------------------
__global__ void embed_tm_kernel(const int* __restrict__ tok, const float* __restrict__ emb,
                                float* __restrict__ out, int Bn, int L, int E) {
  size_t idx = (size_t)blockIdx.x * blockDim.x + threadIdx.x;
  size_t total = (size_t)Bn * L * E;
  if (idx >= total) return;
  int e = (int)(idx % E);
  size_t row = idx / E;
  int b = (int)(row % Bn);
  int l = (int)(row / Bn);
  out[idx] = emb[(size_t)tok[b * L + l] * E + e];
}

// out[r*2H + j] = A[r*H + j]; out[r*2H + H + j] = B[r*H + j]
__global__ void concat2_kernel(const float* __restrict__ A, const float* __restrict__ Bp,
                               float* __restrict__ out, int rows, int Hd) {
  size_t idx = (size_t)blockIdx.x * blockDim.x + threadIdx.x;
  size_t total = (size_t)rows * Hd;
  if (idx >= total) return;
  int j = (int)(idx % Hd);
  size_t r = idx / Hd;
  out[r * (size_t)(2 * Hd) + j]      = A[idx];
  out[r * (size_t)(2 * Hd) + Hd + j] = Bp[idx];
}

__global__ void fill_zero_kernel(float* __restrict__ p, size_t n) {
  size_t i = (size_t)blockIdx.x * blockDim.x + threadIdx.x;
  if (i < n) p[i] = 0.0f;
}

__device__ __forceinline__ float sigmoidf_(float x) { return 1.0f / (1.0f + expf(-x)); }

// LSTM pointwise gates. g: (B, 4H) pre-activation; c,h: (B,H) state (updated);
// hs (optional): also store h (time-major slice pre-offset by host).
__global__ void lstm_gates_kernel(const float* __restrict__ g, float* __restrict__ c,
                                  float* __restrict__ h, float* __restrict__ hs,
                                  int Bn, int Hd) {
  int idx = blockIdx.x * blockDim.x + threadIdx.x;
  if (idx >= Bn * Hd) return;
  int b = idx / Hd, j = idx % Hd;
  const float* gb = g + (size_t)b * 4 * Hd;
  float ig = sigmoidf_(gb[j]);
  float fg = sigmoidf_(gb[Hd + j]);
  float gg = tanhf(gb[2 * Hd + j]);
  float og = sigmoidf_(gb[3 * Hd + j]);
  float cv = fg * c[idx] + ig * gg;
  c[idx] = cv;
  float hv = og * tanhf(cv);
  h[idx] = hv;
  if (hs) hs[idx] = hv;
}

// Dot-product attention over time-major encoder states enc[(s*B + b)*H2 + j].
// One block per batch element.
__global__ __launch_bounds__(256)
void attention_kernel(const float* __restrict__ h1, const float* __restrict__ enc,
                      float* __restrict__ ctx, int Bn, int Sn, int Hd) {
  __shared__ float red[256];
  __shared__ float sc[128];
  const int b = blockIdx.x, tid = threadIdx.x;
  const float* hb = h1 + (size_t)b * Hd;
  for (int s = 0; s < Sn; ++s) {
    const float* eb = enc + ((size_t)s * Bn + b) * Hd;
    float p = 0.0f;
    for (int j = tid; j < Hd; j += 256) p += hb[j] * eb[j];
    red[tid] = p;
    __syncthreads();
    for (int off = 128; off > 0; off >>= 1) {
      if (tid < off) red[tid] += red[tid + off];
      __syncthreads();
    }
    if (tid == 0) sc[s] = red[0];
    __syncthreads();
  }
  if (tid == 0) {
    float mx = -1e30f;
    for (int s = 0; s < Sn; ++s) mx = fmaxf(mx, sc[s]);
    float sum = 0.0f;
    for (int s = 0; s < Sn; ++s) { float e = expf(sc[s] - mx); sc[s] = e; sum += e; }
    float inv = 1.0f / sum;
    for (int s = 0; s < Sn; ++s) sc[s] *= inv;
  }
  __syncthreads();
  for (int j = tid; j < Hd; j += 256) {
    float a = 0.0f;
    for (int s = 0; s < Sn; ++s) a += sc[s] * enc[((size_t)s * Bn + b) * Hd + j];
    ctx[(size_t)b * Hd + j] = a;
  }
}

// ---------------------------------------------------------------------------
// Host-side orchestration helpers
// ---------------------------------------------------------------------------
static inline void launch_gemm(const float* A, const float* W, const float* bias,
                               const float* Cin, float* Cout, int M, int N, int K,
                               int permMod, hipStream_t s) {
  dim3 grid(N / 128, M / 32);
  gemm_wmma_f16<<<grid, 128, 0, s>>>(A, W, bias, Cin, Cout, M, N, K, permMod);
}
static inline void launch_fill0(float* p, size_t n, hipStream_t s) {
  fill_zero_kernel<<<(unsigned)((n + 255) / 256), 256, 0, s>>>(p, n);
}
static inline void launch_concat2(const float* A, const float* Bp, float* out,
                                  int rows, int Hd, hipStream_t s) {
  size_t total = (size_t)rows * Hd;
  concat2_kernel<<<(unsigned)((total + 255) / 256), 256, 0, s>>>(A, Bp, out, rows, Hd);
}
static inline void launch_gates(const float* g, float* c, float* h, float* hs,
                                int Bn, int Hd, hipStream_t s) {
  int total = Bn * Hd;
  lstm_gates_kernel<<<(total + 255) / 256, 256, 0, s>>>(g, c, h, hs, Bn, Hd);
}

extern "C" void kernel_launch(void* const* d_in, const int* in_sizes, int n_in,
                              void* d_out, int out_size, void* d_ws, size_t ws_size,
                              hipStream_t stream) {
  (void)in_sizes; (void)n_in; (void)out_size; (void)ws_size;

  // ---- inputs (setup_inputs dict order, nested lists flattened) ----
  const int*   src     = (const int*)d_in[0];     // (B,S)
  const int*   tgt     = (const int*)d_in[1];     // (B,T)
  const float* emb_src = (const float*)d_in[2];   // (VS,E)
  const float* emb_tgt = (const float*)d_in[3];   // (VT,E)
  // enc_params: layer l (base 4 + 8l): wf, uf, bf, cf, wb, ub, bb, cb
  // dec_params: layer l (base 20 + 4l): w, u, b, d
  const float* ac_w = (const float*)d_in[28];     // (H2, 2H2)
  const float* ac_b = (const float*)d_in[29];     // (H2,)
  const float* op_w = (const float*)d_in[30];     // (VT, H2)
  const float* op_b = (const float*)d_in[31];     // (VT,)
  float* out = (float*)d_out;                     // (B,T,VT)

  // ---- workspace carve-up (floats) ----
  float* ws = (float*)d_ws;
  const size_t M1 = 1u << 20;
  float* X0   = ws;               // 2M : embedded src (time-major) / embedded tgt
  float* X1   = ws + 2 * M1;      // 2M : layer-1 input (concat of layer-0 outputs)
  float* GXF  = ws + 4 * M1;      // 4M : fwd input-projection (also decoder GX0 spans GXF+GXB)
  float* GXB  = ws + 8 * M1;      // 4M : bwd input-projection
  float* HSF  = ws + 12 * M1;     // 1M : fwd hidden sequence (later reused as COMB)
  float* HSB  = ws + 13 * M1;     // 1M : bwd hidden sequence
  float* EO   = ws + 14 * M1;     // 2M : encoder output, time-major (S*B, H2)
  float* SM   = ws + 16 * M1;     // small buffers
  float* hf_s = SM;               // B*H
  float* cf_s = SM + 16384;
  float* hb_s = SM + 32768;
  float* cb_s = SM + 49152;
  float* H0   = SM + 65536;       // B*H2 decoder states
  float* C0   = SM + 98304;
  float* H1   = SM + 131072;
  float* C1   = SM + 163840;
  float* GBUF = SM + 196608;      // B*G2
  float* CTX  = SM + 327680;      // B*H2
  float* CMBI = SM + 360448;      // B*2H2
  float* GX0  = GXF;              // decoder input projection, T*B*G2 (6.29M floats)
  float* COMB = HSF;              // T*B*H2 (1.57M floats)

  // =========================== encoder ===========================
  {
    size_t total = (size_t)Bb * Ss * Ee;
    embed_tm_kernel<<<(unsigned)((total + 255) / 256), 256, 0, stream>>>(
        src, emb_src, X0, Bb, Ss, Ee);
  }
  const float* layer_in = X0;
  int din = Ee;
  for (int layer = 0; layer < 2; ++layer) {
    const float* wf = (const float*)d_in[4 + 8 * layer + 0];
    const float* uf = (const float*)d_in[4 + 8 * layer + 1];
    const float* bf = (const float*)d_in[4 + 8 * layer + 2];
    const float* cf = (const float*)d_in[4 + 8 * layer + 3];
    const float* wb = (const float*)d_in[4 + 8 * layer + 4];
    const float* ub = (const float*)d_in[4 + 8 * layer + 5];
    const float* bb = (const float*)d_in[4 + 8 * layer + 6];
    const float* cb = (const float*)d_in[4 + 8 * layer + 7];

    // Batched input projections for all timesteps: (S*B, 4H) = X @ W^T + b
    launch_gemm(layer_in, wf, bf, nullptr, GXF, Ss * Bb, Gg, din, 0, stream);
    launch_gemm(layer_in, wb, bb, nullptr, GXB, Ss * Bb, Gg, din, 0, stream);

    launch_fill0(hf_s, 4 * (size_t)Bb * Hh, stream);  // hf,cf,hb,cb contiguous

    // forward scan
    for (int l = 0; l < Ss; ++l) {
      launch_gemm(hf_s, uf, cf, GXF + (size_t)l * Bb * Gg, GBUF, Bb, Gg, Hh, 0, stream);
      launch_gates(GBUF, cf_s, hf_s, HSF + (size_t)l * Bb * Hh, Bb, Hh, stream);
    }
    // backward scan
    for (int l = Ss - 1; l >= 0; --l) {
      launch_gemm(hb_s, ub, cb, GXB + (size_t)l * Bb * Gg, GBUF, Bb, Gg, Hh, 0, stream);
      launch_gates(GBUF, cb_s, hb_s, HSB + (size_t)l * Bb * Hh, Bb, Hh, stream);
    }
    // concat outputs -> next layer input (or encoder output), time-major
    float* lo = (layer == 1) ? EO : X1;
    launch_concat2(HSF, HSB, lo, Ss * Bb, Hh, stream);
    // capture final states for decoder init
    launch_concat2(hf_s, hb_s, (layer == 0) ? H0 : H1, Bb, Hh, stream);
    launch_concat2(cf_s, cb_s, (layer == 0) ? C0 : C1, Bb, Hh, stream);
    layer_in = X1;
    din = H2_;
  }

  // =========================== decoder ===========================
  const float* w0 = (const float*)d_in[20];
  const float* u0 = (const float*)d_in[21];
  const float* b0 = (const float*)d_in[22];
  const float* d0 = (const float*)d_in[23];
  const float* w1 = (const float*)d_in[24];
  const float* u1 = (const float*)d_in[25];
  const float* b1 = (const float*)d_in[26];
  const float* d1 = (const float*)d_in[27];

  {
    size_t total = (size_t)Bb * Tt * Ee;
    embed_tm_kernel<<<(unsigned)((total + 255) / 256), 256, 0, stream>>>(
        tgt, emb_tgt, X0, Bb, Tt, Ee);
  }
  // gx0 = emb_tgt[tgt] @ w0^T + b0, all timesteps at once: (T*B, 4H2)
  launch_gemm(X0, w0, b0, nullptr, GX0, Tt * Bb, G2_, Ee, 0, stream);

  for (int t = 0; t < Tt; ++t) {
    // layer 0
    launch_gemm(H0, u0, d0, GX0 + (size_t)t * Bb * G2_, GBUF, Bb, G2_, H2_, 0, stream);
    launch_gates(GBUF, C0, H0, nullptr, Bb, H2_, stream);
    // layer 1: g = h0@w1^T + b1 + h1@u1^T + d1
    launch_gemm(H0, w1, b1, nullptr, GBUF, Bb, G2_, H2_, 0, stream);
    launch_gemm(H1, u1, d1, GBUF, GBUF, Bb, G2_, H2_, 0, stream);
    launch_gates(GBUF, C1, H1, nullptr, Bb, H2_, stream);
    // attention over encoder output
    attention_kernel<<<Bb, 256, 0, stream>>>(H1, EO, CTX, Bb, Ss, H2_);
    // comb = [h1, ctx] @ ac_w^T + ac_b  -> stored time-major for the big GEMM
    launch_concat2(H1, CTX, CMBI, Bb, H2_, stream);
    launch_gemm(CMBI, ac_w, ac_b, nullptr, COMB + (size_t)t * Bb * H2_,
                Bb, H2_, 2 * H2_, 0, stream);
  }

  // Batched output projection: (T*B, VT) = comb @ op_w^T + op_b, written
  // directly in (B,T,VT) order via the row permutation.
  launch_gemm(COMB, op_w, op_b, nullptr, out, Tt * Bb, VT_, H2_, Bb, stream);
}
